// Glimpse_37847251812878
// MI455X (gfx1250) — matrix operand order repaired and checked by reading, not compiled, not run on hardware
//
#include <hip/hip_runtime.h>

// Glimpse attention for MI455X (gfx1250), single fused kernel, single HBM pass
// over `target` (flash-style online softmax), math on v_wmma_f32_16x16x32_bf16.
//
// Refactor: fold the single query into Wk => scores GEMM S(16x256)@T^T;
// fold softmax weights into the target reduction => u GEMM alpha~(16x32)@T.
// Per round: wave w produces the score tile for step 8*round+w into LDS;
// after a uniform barrier each wave consumes two steps' alpha~ for half the
// e-tiles (waves 0-3: e-tiles 0-7, waves 4-7: 8-15), keeping its own flash
// (m,s) state. Persistent accumulators: 8 C-tiles (64 VGPRs).
// f32->bf16: +0x8000 bias + one v_perm_b32 per pair. Batched load clauses
// for the strided u-GEMM B-tiles; global_prefetch for next round's rows.

#define BATCH 256
#define LSEQ  2048
#define EIN   256     // INPUT_SIZE
#define HIDN  256     // HIDDEN_SIZE
#define NHEAD 8
#define DHEAD 32
#define SCALE 0.17677669529663687f   // 1/sqrt(32)
#define NEG_BIG (-1.0e30f)

typedef __attribute__((ext_vector_type(16))) __bf16 v16bf;
typedef __attribute__((ext_vector_type(2)))  __bf16 v2bf;
typedef __attribute__((ext_vector_type(8)))  float  v8f;

union BfPack { unsigned short u[16]; unsigned int i[8]; v2bf h[8]; v16bf v; };

__device__ __forceinline__ unsigned short f2bf(float f) {
  unsigned int x = __float_as_uint(f);
  x += 0x7FFFu + ((x >> 16) & 1u);          // round-to-nearest-even
  return (unsigned short)(x >> 16);
}
__device__ __forceinline__ float bf2f(unsigned short s) {
  return __uint_as_float(((unsigned int)s) << 16);
}
// pack bf16(a) -> low16, bf16(b) -> high16 : 2x v_add + 1x v_perm_b32
__device__ __forceinline__ unsigned int pkbf(float a, float b) {
  unsigned int ua = __float_as_uint(a) + 0x8000u;   // round to nearest
  unsigned int ub = __float_as_uint(b) + 0x8000u;
  return __builtin_amdgcn_perm(ub, ua, 0x07060302); // {ub[3],ub[2],ua[3],ua[2]}
}

__global__ __launch_bounds__(256) void glimpse_attn(
    const float* __restrict__ query, const float* __restrict__ target,
    const unsigned char* __restrict__ mask,
    const float* __restrict__ Wq_w, const float* __restrict__ Wq_b,
    const float* __restrict__ Wk_w, const float* __restrict__ Wk_b,
    const float* __restrict__ Wv_w, const float* __restrict__ Wv_b,
    const float* __restrict__ Wo_w, const float* __restrict__ Wo_b,
    float* __restrict__ alpha_out, float* __restrict__ out)
{
  __shared__ __align__(16) unsigned short scB[NHEAD][LSEQ];  // 32KB raw scores (bf16)
  __shared__ __align__(16) unsigned short Sbf[16][EIN];      // 8KB q-folded K matrix, rows 8..15 = 0
  __shared__ __align__(16) float us[NHEAD][EIN];             // 8KB u accumulator (reused as q scratch)
  __shared__ float att[HIDN];
  __shared__ float cS[NHEAD];
  __shared__ float mW[4][NHEAD], sW[4][NHEAD], gW[NHEAD][NHEAD];
  __shared__ float mS[NHEAD], rS[NHEAD];
  __shared__ int allmS;

  const int b    = blockIdx.x;
  const int t    = threadIdx.x;
  const int wave = t >> 5;
  const int lane = t & 31;
  const int m16  = lane & 15;
  const int hi   = lane >> 4;     // K-half owned by this lane in WMMA A/B layouts

  float* qrow = &us[0][0];        // scratch, dead before u accumulation
  float* qv   = &us[2][0];

  // ---- phase 0: q = Wq * query_b + bq ------------------------------------
  qrow[t] = query[(long)b * EIN + t];
  if (t == 0) allmS = 1;
  __syncthreads();
  {
    float acc = Wq_b[t];
    for (int e = 0; e < EIN; ++e) acc += qrow[e] * Wq_w[t * EIN + e];
    qv[t] = acc;
  }
  __syncthreads();

  // ---- phase 1: S[h,:] = SCALE * q_h . Wk_head (bf16, padded to 16 rows) -
  for (int h = 0; h < NHEAD; ++h) {
    float s = 0.f;
    for (int d = 0; d < DHEAD; ++d) s += qv[h * DHEAD + d] * Wk_w[(h * DHEAD + d) * EIN + t];
    Sbf[h][t]     = f2bf(s * SCALE);
    Sbf[h + 8][t] = 0;
  }
  if (t < NHEAD) {
    float c = 0.f;
    for (int d = 0; d < DHEAD; ++d) c += qv[t * DHEAD + d] * Wk_b[t * DHEAD + d];
    cS[t] = c * SCALE;
  }
  {
    const unsigned char* mp = mask + (long)b * LSEQ;
    int anyzero = 0;
    for (int i = t; i < LSEQ; i += 256) anyzero |= (mp[i] == 0);
    if (anyzero) allmS = 0;
  }
  __syncthreads();

  // zero the u accumulator (q scratch is dead now)
  { float* uf = &us[0][0]; for (int i = t; i < NHEAD * EIN; i += 256) uf[i] = 0.f; }
  const bool allm = (allmS != 0);
  __syncthreads();

  // ---- main loop: 8 rounds; produce 1 step (32 keys), consume 2 steps ----
  float rowm = NEG_BIG, rowsum = 0.f;     // per-lane flash state for row m16 (valid m16 < 8)
  v8f acc[8];
#pragma unroll
  for (int j = 0; j < 8; ++j) acc[j] = (v8f){};

  const int cw     = wave & 3;            // consumer step phase within a round
  const int etbase = (wave < 4) ? 0 : 8;  // consumer e-tile half
  const long tbase = (long)b * LSEQ;

  for (int round = 0; round < 8; ++round) {
    // prefetch next round's producer rows (each lane: one row, 8 x 128B lines)
    if (round < 7) {
      const float* pf = target + (tbase + ((round + 1) * 8 + wave) * 32 + lane) * EIN;
#pragma unroll
      for (int pq = 0; pq < 8; ++pq) __builtin_prefetch(pf + pq * 32, 0, 1);
    }

    // ======== produce: scores C(16x32) = S @ T^T for step sp ========
    const int lp = (round * 8 + wave) * 32;
    {
      v8f s0 = {}, s1 = {};
      const float* tp0 = target + (tbase + lp + m16) * EIN + hi * 16;
      const float* tp1 = tp0 + 16 * EIN;
      const int koff = hi ? 16 : 0;
#pragma unroll
      for (int k = 0; k < 8; ++k) {
        BfPack a;
        *(uint4*)&a.u[0] = *(const uint4*)&Sbf[m16][k * 32 + koff];
        *(uint4*)&a.u[8] = *(const uint4*)&Sbf[m16][k * 32 + koff + 8];
        BfPack b0, b1;
#pragma unroll
        for (int j = 0; j < 16; j += 4) {
          float4 f = *(const float4*)(tp0 + k * 32 + j);
          b0.i[j / 2]     = pkbf(f.x, f.y);
          b0.i[j / 2 + 1] = pkbf(f.z, f.w);
          float4 g = *(const float4*)(tp1 + k * 32 + j);
          b1.i[j / 2]     = pkbf(g.x, g.y);
          b1.i[j / 2 + 1] = pkbf(g.z, g.w);
        }
        s0 = __builtin_amdgcn_wmma_f32_16x16x32_bf16(false, a.v, false, b0.v,
                                                     (short)0, s0, false, false);
        s1 = __builtin_amdgcn_wmma_f32_16x16x32_bf16(false, a.v, false, b1.v,
                                                     (short)0, s1, false, false);
      }
      // bias + mask, raw bf16 scores -> LDS (C layout: lanes 0-15 carry rows 0..7)
      if (lane < 16) {
        const unsigned char mk0 = mask[tbase + lp + lane];
        const unsigned char mk1 = mask[tbase + lp + 16 + lane];
#pragma unroll
        for (int r = 0; r < NHEAD; ++r) {
          float x0 = s0[r] + cS[r]; x0 = mk0 ? -10000.0f : x0; if (allm) x0 = 0.f;
          float x1 = s1[r] + cS[r]; x1 = mk1 ? -10000.0f : x1; if (allm) x1 = 0.f;
          scB[r][lp + lane]      = f2bf(x0);
          scB[r][lp + 16 + lane] = f2bf(x1);
        }
      }
    }
    __syncthreads();   // uniform: round's scores visible to consumers

    // ======== consume: 2 steps of alpha~, half of the e-tiles ========
#pragma unroll
    for (int ss = 0; ss < 2; ++ss) {
      const int lc = (round * 8 + cw + ss * 4) * 32;
      // A-layout read (lane m16 owns row m16; hi selects K-half)
      float svals[16];
      float tmax = NEG_BIG;
      const int ka = lc + (hi ? 16 : 0);
      if (m16 < NHEAD) {
        BfPack p;
        *(uint4*)&p.u[0] = *(const uint4*)&scB[m16][ka];
        *(uint4*)&p.u[8] = *(const uint4*)&scB[m16][ka + 8];
#pragma unroll
        for (int j = 0; j < 16; ++j) { svals[j] = bf2f(p.u[j]); tmax = fmaxf(tmax, svals[j]); }
      }
      tmax = fmaxf(tmax, __shfl_xor(tmax, 16));
      const float mnew   = fmaxf(rowm, tmax);
      const float fscale = __expf(rowm - mnew);
      BfPack pal;
      float lsum = 0.f;
      if (m16 < NHEAD) {
#pragma unroll
        for (int j = 0; j < 16; j += 2) {
          float e0 = __expf(svals[j] - mnew);
          float e1 = __expf(svals[j + 1] - mnew);
          lsum += e0 + e1;
          pal.i[j / 2] = pkbf(e0, e1);
        }
      } else {
#pragma unroll
        for (int j = 0; j < 8; ++j) pal.i[j] = 0;
      }
      lsum += __shfl_xor(lsum, 16);
      rowsum = rowsum * fscale + lsum;
      rowm   = mnew;
      float f_all[NHEAD];
#pragma unroll
      for (int r = 0; r < NHEAD; ++r) f_all[r] = __shfl(fscale, r);

      // u update: acc_j = diag(f) * acc_j + alpha~ (16x32) @ T (32x16)
      const float* tb0 = target + (tbase + lc + (hi ? 16 : 0)) * EIN;
#pragma unroll
      for (int j = 0; j < 8; ++j) {
        const float* tb = tb0 + (etbase + j) * 16 + m16;
        float vals[16];                     // keep all 16 loads in flight (one clause)
#pragma unroll
        for (int q = 0; q < 16; ++q) vals[q] = tb[(long)q * EIN];
        BfPack pb;
#pragma unroll
        for (int q = 0; q < 16; q += 2) pb.i[q / 2] = pkbf(vals[q], vals[q + 1]);
        v8f c = acc[j];
#pragma unroll
        for (int r = 0; r < NHEAD; ++r) c[r] *= f_all[r];
        acc[j] = __builtin_amdgcn_wmma_f32_16x16x32_bf16(false, pal.v, false, pb.v,
                                                         (short)0, c, false, false);
      }
    }
  }

  // ---- merge per-consumer (m, s, u); waves w and w+4 hold identical (m,s) -
  if (wave < 4 && lane < NHEAD) { mW[wave][lane] = rowm; sW[wave][lane] = rowsum; }
  __syncthreads();
  if (t < NHEAD) {
    float ms = NEG_BIG;
#pragma unroll
    for (int w = 0; w < 4; ++w) ms = fmaxf(ms, mW[w][t]);
    float ss = 0.f;
#pragma unroll
    for (int w = 0; w < 4; ++w) ss += sW[w][t] * __expf(mW[w][t] - ms);
    mS[t] = ms;
    rS[t] = 1.0f / ss;
  }
  __syncthreads();
  if (t < 64) {
    const int w = t >> 3, h2 = t & 7;
    gW[w][h2] = __expf(mW[w & 3][h2] - mS[h2]) * rS[h2];
  }
  __syncthreads();
  {
    float g[NHEAD];
#pragma unroll
    for (int r = 0; r < NHEAD; ++r) g[r] = gW[wave][r];
    if (lane < 16) {
#pragma unroll
      for (int j = 0; j < 8; ++j)
#pragma unroll
        for (int r = 0; r < NHEAD; ++r)
          atomicAdd(&us[r][(etbase + j) * 16 + lane], acc[j][r] * g[r]);
    }
  }
  // alpha output from raw bf16 scores: wave w owns head row w
  {
    const int h = wave;
    const float ms = mS[h], inv = rS[h];
    float* aout = alpha_out + ((long)b * NHEAD + h) * LSEQ;
    for (int l = lane; l < LSEQ; l += 32)
      aout[l] = __expf(bf2f(scB[h][l]) - ms) * inv;
  }
  __syncthreads();

  // ---- epilogue: attended = Wv @ u + bv ; out = Wo @ attended + bo -------
  {
    const int h = t >> 5;
    float a = Wv_b[t];
    for (int e = 0; e < EIN; ++e) a += Wv_w[t * EIN + e] * us[h][e];
    att[t] = a;
  }
  __syncthreads();
  {
    float r = Wo_b[t];
    for (int i = 0; i < HIDN; ++i) r += Wo_w[t * HIDN + i] * att[i];
    out[(long)b * EIN + t] = r;
  }
}

extern "C" void kernel_launch(void* const* d_in, const int* in_sizes, int n_in,
                              void* d_out, int out_size, void* d_ws, size_t ws_size,
                              hipStream_t stream) {
  const float*         query  = (const float*)d_in[0];
  const float*         target = (const float*)d_in[1];
  const unsigned char* mask   = (const unsigned char*)d_in[2];
  const float* Wq_w = (const float*)d_in[3];
  const float* Wq_b = (const float*)d_in[4];
  const float* Wk_w = (const float*)d_in[5];
  const float* Wk_b = (const float*)d_in[6];
  const float* Wv_w = (const float*)d_in[7];
  const float* Wv_b = (const float*)d_in[8];
  const float* Wo_w = (const float*)d_in[9];
  const float* Wo_b = (const float*)d_in[10];

  float* alpha_out = (float*)d_out;                           // (B, 8, 2048)
  float* out       = alpha_out + (long)BATCH * NHEAD * LSEQ;  // (B, 256)

  glimpse_attn<<<BATCH, 256, 0, stream>>>(query, target, mask,
                                          Wq_w, Wq_b, Wk_w, Wk_b,
                                          Wv_w, Wv_b, Wo_w, Wo_b,
                                          alpha_out, out);
}
